// SimilarityDrivenVectorQuantizer_1047972020229
// MI455X (gfx1250) — matrix operand order repaired and checked by reading, not compiled, not run on hardware
//
#include <hip/hip_runtime.h>
#include <hip/hip_bf16.h>

// ---- problem constants (from the reference) ----
#define NUM_EMB   1024
#define EMB_DIM   64
#define BATCH     32
#define TLEN      4096
#define ROWS      256          // t-values (rows) per workgroup
#define NTHREADS  256          // 8 waves (wave32), 32 rows per wave (2 M-tiles)
#define A_STRIDE  72           // bf16 row stride in LDS (16B-aligned, padded)

typedef __bf16        v16bf __attribute__((ext_vector_type(16)));
typedef float         v8f   __attribute__((ext_vector_type(8)));
typedef unsigned int  v4u   __attribute__((ext_vector_type(4)));
typedef unsigned int  v8u   __attribute__((ext_vector_type(8)));

__device__ __forceinline__ unsigned short f32_to_bf16_rne(float f) {
  unsigned int u = __builtin_bit_cast(unsigned int, f);
  unsigned int r = u + 0x7FFFu + ((u >> 16) & 1u);   // round-to-nearest-even
  return (unsigned short)(r >> 16);
}

__device__ __forceinline__ v16bf make_frag(const unsigned short* p0,
                                           const unsigned short* p1) {
  v8u q;
  q.lo = *(const v4u*)p0;   // 8 bf16 (16B)
  q.hi = *(const v4u*)p1;   // 8 bf16 (16B)
  return __builtin_bit_cast(v16bf, q);
}

// ---- kernel 0: zero the loss accumulator in d_ws (graph-replay safe) ----
__global__ void vq_zero_ws(float* ws) {
  if (threadIdx.x == 0) ws[0] = 0.0f;
}

// ---- kernel 1: fp32 codebook -> bf16 (row-major [code][d]) in d_ws ----
__global__ void vq_convert_codebook(const float* __restrict__ emb,
                                    unsigned short* __restrict__ cb) {
  int i = blockIdx.x * blockDim.x + threadIdx.x;
  if (i < NUM_EMB * EMB_DIM) cb[i] = f32_to_bf16_rne(emb[i]);
}

// ---- kernel 2: fused normalize + WMMA similarity + argmax + gather + loss ----
__launch_bounds__(NTHREADS)
__global__ void vq_main(const float* __restrict__ inputs,          // [B,D,T]
                        const float* __restrict__ embedding,       // [K,D] normalized
                        const float* __restrict__ emb_unnorm,      // [K,D]
                        const unsigned short* __restrict__ cb,     // [K,D] bf16
                        float* __restrict__ out,                   // concat outputs
                        float* __restrict__ loss_accum) {
  __shared__ __align__(16) unsigned short s_a[ROWS * A_STRIDE];    // normalized bf16 rows
  __shared__ int s_idx[ROWS];

  const int tid  = threadIdx.x;
  const int lane = tid & 31;
  const int wave = tid >> 5;

  const int blocksPerB = TLEN / ROWS;                  // 16
  const int b  = blockIdx.x / blocksPerB;
  const int t0 = (blockIdx.x % blocksPerB) * ROWS;
  const float* xin = inputs + (size_t)b * EMB_DIM * TLEN + t0;

  // --- stage 1: per-row L2 norm + bf16 normalized rows into LDS ---
  // lanes index consecutive t -> every global load is coalesced.
  {
    float ss = 0.0f;
#pragma unroll 8
    for (int d = 0; d < EMB_DIM; ++d) {
      float v = xin[(size_t)d * TLEN + tid];
      ss += v * v;
    }
    float rinv = rsqrtf(ss);
#pragma unroll 8
    for (int d = 0; d < EMB_DIM; ++d) {
      float v = xin[(size_t)d * TLEN + tid];
      s_a[tid * A_STRIDE + d] = f32_to_bf16_rne(v * rinv);
    }
  }
  __syncthreads();

  // --- stage 2: per-wave 2x16-row M-tiles vs all 1024 codes, argmax in-register ---
  const int m0 = wave * 32;         // 8 waves x 32 rows = 256 rows
  const int hi = lane >> 4;         // half-wave select
  const int lc = lane & 15;

  // A fragments, ISA 16-bit A 16x32 layout:
  //   lanes 0-15: K {0..7, 16..23};  lanes 16-31: K {8..15, 24..31}  (+32 for k-step 1)
  const unsigned short* ap0 = &s_a[(m0 + lc) * A_STRIDE];
  const unsigned short* ap1 = &s_a[(m0 + 16 + lc) * A_STRIDE];
  v16bf a00 = make_frag(ap0 + hi * 8,      ap0 + hi * 8 + 16);   // tile0, k-step 0
  v16bf a01 = make_frag(ap0 + 32 + hi * 8, ap0 + 32 + hi * 8 + 16);
  v16bf a10 = make_frag(ap1 + hi * 8,      ap1 + hi * 8 + 16);   // tile1, k-step 0
  v16bf a11 = make_frag(ap1 + 32 + hi * 8, ap1 + 32 + hi * 8 + 16);

  float bv0[8], bv1[8];
  int   bi0[8], bi1[8];
#pragma unroll
  for (int i = 0; i < 8; ++i) {
    bv0[i] = -3.4e38f; bi0[i] = 0;
    bv1[i] = -3.4e38f; bi1[i] = 0;
  }

#pragma unroll 2
  for (int nc = 0; nc < NUM_EMB / 16; ++nc) {
    const int n = nc * 16 + lc;                        // this lane's column (code id)
    const unsigned short* bp = cb + n * EMB_DIM;
    // B fragments, same K order as A: lanes 0-15 hold K {0..7, 16..23}, lanes 16-31 hold K {8..15, 24..31}
    v16bf b0 = make_frag(bp + hi * 8,      bp + hi * 8 + 16);
    v16bf b1 = make_frag(bp + 32 + hi * 8, bp + 32 + hi * 8 + 16);
    if (nc + 4 < NUM_EMB / 16)
      __builtin_prefetch(bp + 4 * 16 * EMB_DIM, 0, 1);   // global_prefetch_b8

    // two independent accumulation chains share each B fragment
    v8f c0 = {};
    v8f c1 = {};
    c0 = __builtin_amdgcn_wmma_f32_16x16x32_bf16(false, a00, false, b0,
                                                 (short)0, c0, false, false);
    c1 = __builtin_amdgcn_wmma_f32_16x16x32_bf16(false, a10, false, b0,
                                                 (short)0, c1, false, false);
    c0 = __builtin_amdgcn_wmma_f32_16x16x32_bf16(false, a01, false, b1,
                                                 (short)0, c0, false, false);
    c1 = __builtin_amdgcn_wmma_f32_16x16x32_bf16(false, a11, false, b1,
                                                 (short)0, c1, false, false);
    // C layout: VGPR i <-> row (tile_base + i + 8*hi) ; lane <-> column n
#pragma unroll
    for (int i = 0; i < 8; ++i) {
      if (c0[i] > bv0[i]) { bv0[i] = c0[i]; bi0[i] = n; }  // ascending n -> first max wins
      if (c1[i] > bv1[i]) { bv1[i] = c1[i]; bi1[i] = n; }
    }
  }

  // argmax reduction across the 16 lanes of each half-wave (first-index tie-break)
#pragma unroll
  for (int off = 1; off < 16; off <<= 1) {
#pragma unroll
    for (int i = 0; i < 8; ++i) {
      float ov0 = __shfl_xor(bv0[i], off, 32);
      int   oi0 = __shfl_xor(bi0[i], off, 32);
      if (ov0 > bv0[i] || (ov0 == bv0[i] && oi0 < bi0[i])) { bv0[i] = ov0; bi0[i] = oi0; }
      float ov1 = __shfl_xor(bv1[i], off, 32);
      int   oi1 = __shfl_xor(bi1[i], off, 32);
      if (ov1 > bv1[i] || (ov1 == bv1[i] && oi1 < bi1[i])) { bv1[i] = ov1; bi1[i] = oi1; }
    }
  }
  if (lc == 0) {
#pragma unroll
    for (int i = 0; i < 8; ++i) {
      s_idx[m0 + hi * 8 + i]      = bi0[i];
      s_idx[m0 + 16 + hi * 8 + i] = bi1[i];
    }
  }
  __syncthreads();

  // --- stage 3: outputs ---
  const size_t QELEMS = (size_t)BATCH * EMB_DIM * TLEN;   // 8388608
  float* outq   = out;
  float* outidx = out + QELEMS + 2;

  const int t    = tid;                                   // one row per thread
  const int code = s_idx[t];

  // quantized: out[b, d, t0+t] = embedding[code, d]  (stores coalesced along t)
  {
    const float4* er = (const float4*)(embedding + (size_t)code * EMB_DIM);
    float* op = outq + (size_t)b * EMB_DIM * TLEN + t0 + t;
#pragma unroll 4
    for (int dc = 0; dc < EMB_DIM / 4; ++dc) {
      float4 q = er[dc];
      int d = dc * 4;
      op[(size_t)(d + 0) * TLEN] = q.x;
      op[(size_t)(d + 1) * TLEN] = q.y;
      op[(size_t)(d + 2) * TLEN] = q.z;
      op[(size_t)(d + 3) * TLEN] = q.w;
    }
  }
  // idx (stored as float; exact for idx < 2^24)
  outidx[(size_t)b * TLEN + t0 + t] = (float)code;

  // loss partial: sum_d (x[t,d] - emb_unnorm[code,d])^2   (both losses share this value)
  float part = 0.0f;
  {
    const float4* eu = (const float4*)(emb_unnorm + (size_t)code * EMB_DIM);
#pragma unroll 4
    for (int dc = 0; dc < EMB_DIM / 4; ++dc) {
      int d = dc * 4;
      float4 e = eu[dc];
      float x0 = xin[(size_t)(d + 0) * TLEN + t] - e.x;
      float x1 = xin[(size_t)(d + 1) * TLEN + t] - e.y;
      float x2 = xin[(size_t)(d + 2) * TLEN + t] - e.z;
      float x3 = xin[(size_t)(d + 3) * TLEN + t] - e.w;
      part += x0 * x0 + x1 * x1 + x2 * x2 + x3 * x3;
    }
  }
#pragma unroll
  for (int off = 1; off < 32; off <<= 1) part += __shfl_xor(part, off, 32);
  if (lane == 0) atomicAdd(loss_accum, part);
}

// ---- kernel 3: finalize losses ----
__global__ void vq_finish(const float* __restrict__ ws, float* __restrict__ out) {
  if (threadIdx.x == 0) {
    const size_t Q = (size_t)BATCH * EMB_DIM * TLEN;
    float mean = ws[0] / (float)((size_t)BATCH * TLEN * EMB_DIM);
    out[Q]     = mean;   // commitment_loss
    out[Q + 1] = mean;   // codebook_loss (same numeric value; diff is gradient-only)
  }
}

extern "C" void kernel_launch(void* const* d_in, const int* in_sizes, int n_in,
                              void* d_out, int out_size, void* d_ws, size_t ws_size,
                              hipStream_t stream) {
  const float* inputs    = (const float*)d_in[0];   // [32,64,4096]
  const float* embedding = (const float*)d_in[1];   // [1024,64] normalized
  const float* emb_un    = (const float*)d_in[2];   // [1024,64]
  float*       out       = (float*)d_out;

  float*          ws_loss = (float*)d_ws;
  unsigned short* cb_bf16 = (unsigned short*)((char*)d_ws + 256);  // 128 KB, L2-resident

  vq_zero_ws<<<1, 32, 0, stream>>>(ws_loss);
  vq_convert_codebook<<<(NUM_EMB * EMB_DIM + 255) / 256, 256, 0, stream>>>(embedding, cb_bf16);
  vq_main<<<BATCH * (TLEN / ROWS), NTHREADS, 0, stream>>>(inputs, embedding, emb_un,
                                                          cb_bf16, out, ws_loss);
  vq_finish<<<1, 32, 0, stream>>>(ws_loss, out);
}
